// MyIN_88338887344146
// MI455X (gfx1250) — matrix operand order repaired and checked by reading, not compiled, not run on hardware
//
#include <hip/hip_runtime.h>
#include <hip/hip_bf16.h>
#include <math.h>

typedef __attribute__((ext_vector_type(16))) _Float16 v16h;
typedef __attribute__((ext_vector_type(8)))  float    v8f;

#define WMMA_F16(a, b, c) \
  __builtin_amdgcn_wmma_f32_16x16x32_f16(false, (a), false, (b), (short)0, (c), false, false)

// ---------------------------------------------------------------------------
// Packed-weight layout (per MLP, byte offsets inside an 11776-byte region):
//   W1p f16 [32 x 48] @ 0      (K padded 32, N padded 48)
//   W2p f16 [64 x 48] @ 3072
//   W3p f16 [64 x 16] @ 9216
//   b1p f32 [48]      @ 11264
//   b2p f32 [48]      @ 11456
//   b3p f32 [16]      @ 11648
// ---------------------------------------------------------------------------
#define WPACK_STRIDE 11776

// B fragment (K=32 chunk, N-tile of 16) from row-major f16 [Kpad x npad] in LDS.
__device__ __forceinline__ v16h load_B_frag(const _Float16* w, int npad, int kchunk, int ntile) {
  const int lane = threadIdx.x & 31;
  const int n  = ntile * 16 + (lane & 15);
  const int k0 = kchunk * 32 + ((lane >> 4) << 3);
  v16h b;
#pragma unroll
  for (int i = 0; i < 16; ++i) b[i] = w[(k0 + (i < 8 ? i : i + 8)) * npad + n];
  return b;
}

// A fragment (16 rows x K=32 chunk) from row-major f16 [16 x 64] staging in LDS.
__device__ __forceinline__ v16h load_A_frag(const _Float16* A, int kchunk) {
  const int lane = threadIdx.x & 31;
  const int m  = lane & 15;
  const int hi = (lane >> 4) << 3;  // 0 or 8
  const _Float16* row = A + m * 64 + kchunk * 32;
  v16h a;
#pragma unroll
  for (int i = 0; i < 8; ++i) a[i] = row[hi + i];
#pragma unroll
  for (int i = 0; i < 8; ++i) a[8 + i] = row[16 + hi + i];
  return a;
}

// Store a 16x16 f32 D fragment (bias + optional relu) into staging as f16.
__device__ __forceinline__ void store_h_frag(_Float16* A, int ntile, v8f acc,
                                             const float* bias, bool relu) {
  const int lane = threadIdx.x & 31;
  const int n  = ntile * 16 + (lane & 15);
  const int mb = (lane >> 4) << 3;
  const float bn = bias[n];
#pragma unroll
  for (int i = 0; i < 8; ++i) {
    float v = acc[i] + bn;
    if (relu) v = fmaxf(v, 0.0f);
    A[(mb + i) * 64 + n] = (_Float16)v;
  }
}

#define LDS_FENCE() asm volatile("" ::: "memory")

// ---------------------------------------------------------------------------
// Weight packing: fp32 params -> padded f16 B-matrices + fp32 biases
// ---------------------------------------------------------------------------
struct WPtrs { const float* p[42]; };

__global__ void pack_weights(WPtrs w, unsigned char* wpack) {
  const int dins[7]  = {9, 9, 18, 14, 24, 17, 27};
  const int douts[7] = {6, 6, 8, 8, 9, 9, 1};
  const int tid = threadIdx.x;
  for (int m = 0; m < 7; ++m) {
    unsigned char* base = wpack + (size_t)m * WPACK_STRIDE;
    const int din = dins[m], dout = douts[m];
    const float* W1 = w.p[6 * m + 0];
    const float* b1 = w.p[6 * m + 1];
    const float* W2 = w.p[6 * m + 2];
    const float* b2 = w.p[6 * m + 3];
    const float* W3 = w.p[6 * m + 4];
    const float* b3 = w.p[6 * m + 5];
    _Float16* W1p = (_Float16*)(base);
    _Float16* W2p = (_Float16*)(base + 3072);
    _Float16* W3p = (_Float16*)(base + 9216);
    float* b1p = (float*)(base + 11264);
    float* b2p = (float*)(base + 11456);
    float* b3p = (float*)(base + 11648);
    for (int i = tid; i < 32 * 48; i += blockDim.x) {
      const int k = i / 48, n = i - k * 48;
      W1p[i] = (_Float16)((k < din && n < 40) ? W1[k * 40 + n] : 0.0f);
    }
    for (int i = tid; i < 64 * 48; i += blockDim.x) {
      const int k = i / 48, n = i - k * 48;
      W2p[i] = (_Float16)((k < 40 && n < 40) ? W2[k * 40 + n] : 0.0f);
    }
    for (int i = tid; i < 64 * 16; i += blockDim.x) {
      const int k = i / 16, n = i - k * 16;
      W3p[i] = (_Float16)((k < 40 && n < dout) ? W3[k * dout + n] : 0.0f);
    }
    for (int i = tid; i < 48; i += blockDim.x) {
      b1p[i] = (i < 40) ? b1[i] : 0.0f;
      b2p[i] = (i < 40) ? b2[i] : 0.0f;
    }
    for (int i = tid; i < 16; i += blockDim.x) b3p[i] = (i < dout) ? b3[i] : 0.0f;
  }
}

// ---------------------------------------------------------------------------
// Shared per-kernel weight staging
// ---------------------------------------------------------------------------
#define DECLARE_MLP_LDS()            \
  __shared__ _Float16 sW1[32 * 48];  \
  __shared__ _Float16 sW2[64 * 48];  \
  __shared__ _Float16 sW3[64 * 16];  \
  __shared__ float sB1[48], sB2[48], sB3[16];

#define LOAD_MLP_LDS(wp)                                                        \
  do {                                                                          \
    const int t_ = threadIdx.x;                                                 \
    const _Float16* g1_ = (const _Float16*)(wp);                                \
    for (int i = t_; i < 32 * 48; i += 128) sW1[i] = g1_[i];                    \
    const _Float16* g2_ = (const _Float16*)((wp) + 3072);                       \
    for (int i = t_; i < 64 * 48; i += 128) sW2[i] = g2_[i];                    \
    const _Float16* g3_ = (const _Float16*)((wp) + 9216);                       \
    for (int i = t_; i < 64 * 16; i += 128) sW3[i] = g3_[i];                    \
    const float* gb1_ = (const float*)((wp) + 11264);                           \
    const float* gb2_ = (const float*)((wp) + 11456);                           \
    const float* gb3_ = (const float*)((wp) + 11648);                           \
    if (t_ < 48) { sB1[t_] = gb1_[t_]; sB2[t_] = gb2_[t_]; }                    \
    if (t_ < 16) sB3[t_] = gb3_[t_];                                            \
    __syncthreads();                                                            \
  } while (0)

struct Frags { v16h B1[3], B2_0[3], B2_1[3], B3_0, B3_1; };

#define BUILD_B_FRAGS(F)                               \
  _Pragma("unroll")                                    \
  for (int t = 0; t < 3; ++t) {                        \
    (F).B1[t]   = load_B_frag(sW1, 48, 0, t);          \
    (F).B2_0[t] = load_B_frag(sW2, 48, 0, t);          \
    (F).B2_1[t] = load_B_frag(sW2, 48, 1, t);          \
  }                                                    \
  (F).B3_0 = load_B_frag(sW3, 16, 0, 0);               \
  (F).B3_1 = load_B_frag(sW3, 16, 1, 0);

// Two independent 16-row tiles through the 3-layer MLP, WMMAs interleaved so
// the scheduler can fill WMMA->VALU hazard slots with the other tile's chain.
__device__ __forceinline__ void mlp_tile2(_Float16* A0, _Float16* A1, const Frags& F,
                                          const float* sB1, const float* sB2,
                                          v8f& d0, v8f& d1) {
  v8f h0[3], h1[3];
  {  // layer 1: [16 x 32] x [32 x 48]
    v16h a00 = load_A_frag(A0, 0);
    v16h a10 = load_A_frag(A1, 0);
#pragma unroll
    for (int t = 0; t < 3; ++t) {
      { v8f c = {}; h0[t] = WMMA_F16(a00, F.B1[t], c); }
      { v8f c = {}; h1[t] = WMMA_F16(a10, F.B1[t], c); }
    }
    LDS_FENCE();
#pragma unroll
    for (int t = 0; t < 3; ++t) {
      store_h_frag(A0, t, h0[t], sB1, true);
      store_h_frag(A1, t, h1[t], sB1, true);
    }
  }
  LDS_FENCE();
  {  // layer 2: [16 x 64] x [64 x 48]
    v16h a00 = load_A_frag(A0, 0), a01 = load_A_frag(A0, 1);
    v16h a10 = load_A_frag(A1, 0), a11 = load_A_frag(A1, 1);
#pragma unroll
    for (int t = 0; t < 3; ++t) {
      { v8f c = {}; c = WMMA_F16(a00, F.B2_0[t], c); h0[t] = WMMA_F16(a01, F.B2_1[t], c); }
      { v8f c = {}; c = WMMA_F16(a10, F.B2_0[t], c); h1[t] = WMMA_F16(a11, F.B2_1[t], c); }
    }
    LDS_FENCE();
#pragma unroll
    for (int t = 0; t < 3; ++t) {
      store_h_frag(A0, t, h0[t], sB2, true);
      store_h_frag(A1, t, h1[t], sB2, true);
    }
  }
  LDS_FENCE();
  {  // layer 3: [16 x 64] x [64 x 16]
    v16h a00 = load_A_frag(A0, 0), a01 = load_A_frag(A0, 1);
    v16h a10 = load_A_frag(A1, 0), a11 = load_A_frag(A1, 1);
    { v8f c = {}; c = WMMA_F16(a00, F.B3_0, c); d0 = WMMA_F16(a01, F.B3_1, c); }
    { v8f c = {}; c = WMMA_F16(a10, F.B3_0, c); d1 = WMMA_F16(a11, F.B3_1, c); }
  }
}

// ---------------------------------------------------------------------------
// R1 edge kernel: gather [x[dst] | x[src] | e_attr] -> MLP -> E (f16) + atomic aggr
// Each wave processes 32 edges (two 16-row WMMA tiles) per iteration.
// ---------------------------------------------------------------------------
template <int NIN, int EIN, int EOUT, typename TEIN>
__global__ __launch_bounds__(128) void r1_kernel(
    const float* __restrict__ xin, const int* __restrict__ srcIdx,
    const int* __restrict__ dstIdx, const TEIN* __restrict__ eattr,
    const unsigned char* __restrict__ wp, _Float16* __restrict__ Eout,
    float* __restrict__ aggr, int nEdges) {
  constexpr int DIN = 2 * NIN + EIN;
  static_assert(DIN <= 32, "single K chunk");
  DECLARE_MLP_LDS();
  __shared__ _Float16 sA[4][2][16 * 64];
  __shared__ int sDst[4][32];
  LOAD_MLP_LDS(wp);

  const int lane = threadIdx.x & 31;
  const int wave = threadIdx.x >> 5;
  _Float16* A0 = sA[wave][0];
  _Float16* A1 = sA[wave][1];
  int* myDst = sDst[wave];
  Frags F;
  BUILD_B_FRAGS(F);

  const int nPairs = (nEdges + 31) >> 5;
  for (int pair = blockIdx.x * 4 + wave; pair < nPairs; pair += gridDim.x * 4) {
    const int e0 = pair << 5;
#pragma unroll
    for (int i = 0; i < 4; ++i) {
      ((int4*)A0)[lane + 32 * i] = make_int4(0, 0, 0, 0);
      ((int4*)A1)[lane + 32 * i] = make_int4(0, 0, 0, 0);
    }
    {
      const int e = e0 + lane;
      myDst[lane] = (e < nEdges) ? dstIdx[e] : 0;
    }
    LDS_FENCE();
    for (int idx = lane; idx < 32 * DIN; idx += 32) {
      const int r = idx / DIN, c = idx - r * DIN;
      const int e = e0 + r;
      float v = 0.0f;
      if (e < nEdges) {
        if (c < NIN)            v = xin[(size_t)myDst[r] * NIN + c];
        else if (c < 2 * NIN)   v = xin[(size_t)srcIdx[e] * NIN + (c - NIN)];
        else                    v = (float)eattr[(size_t)e * EIN + (c - 2 * NIN)];
      }
      _Float16* A = (r < 16) ? A0 : A1;
      A[(r & 15) * 64 + c] = (_Float16)v;
    }
    LDS_FENCE();
    v8f d0, d1;
    mlp_tile2(A0, A1, F, sB1, sB2, d0, d1);
    const int n = lane & 15;
    const int mb = (lane >> 4) << 3;
    if (n < EOUT) {
      const float bn = sB3[n];
#pragma unroll
      for (int half = 0; half < 2; ++half) {
        const v8f& d = half ? d1 : d0;
        const int rbase = half * 16 + mb;
#pragma unroll
        for (int i = 0; i < 8; ++i) {
          const int r = rbase + i;
          const int e = e0 + r;
          if (e < nEdges) {
            const float v = d[i] + bn;
            Eout[(size_t)e * EOUT + n] = (_Float16)v;
            atomicAdd(&aggr[(size_t)myDst[r] * EOUT + n], v);
          }
        }
      }
    }
  }
}

// ---------------------------------------------------------------------------
// O node kernel: [x | aggr] -> MLP -> new node features (f32)
// ---------------------------------------------------------------------------
template <int NIN, int EOUT, int NOUT>
__global__ __launch_bounds__(128) void o_kernel(
    const float* __restrict__ xin, const float* __restrict__ aggr,
    const unsigned char* __restrict__ wp, float* __restrict__ xout, int nNodes) {
  constexpr int DIN = NIN + EOUT;
  static_assert(DIN <= 32, "single K chunk");
  DECLARE_MLP_LDS();
  __shared__ _Float16 sA[4][2][16 * 64];
  LOAD_MLP_LDS(wp);

  const int lane = threadIdx.x & 31;
  const int wave = threadIdx.x >> 5;
  _Float16* A0 = sA[wave][0];
  _Float16* A1 = sA[wave][1];
  Frags F;
  BUILD_B_FRAGS(F);

  const int nPairs = (nNodes + 31) >> 5;
  for (int pair = blockIdx.x * 4 + wave; pair < nPairs; pair += gridDim.x * 4) {
    const int n0 = pair << 5;
#pragma unroll
    for (int i = 0; i < 4; ++i) {
      ((int4*)A0)[lane + 32 * i] = make_int4(0, 0, 0, 0);
      ((int4*)A1)[lane + 32 * i] = make_int4(0, 0, 0, 0);
    }
    LDS_FENCE();
    for (int idx = lane; idx < 32 * DIN; idx += 32) {
      const int r = idx / DIN, c = idx - r * DIN;
      const int node = n0 + r;
      float v = 0.0f;
      if (node < nNodes) {
        if (c < NIN) v = xin[(size_t)node * NIN + c];
        else         v = aggr[(size_t)node * EOUT + (c - NIN)];
      }
      _Float16* A = (r < 16) ? A0 : A1;
      A[(r & 15) * 64 + c] = (_Float16)v;
    }
    LDS_FENCE();
    v8f d0, d1;
    mlp_tile2(A0, A1, F, sB1, sB2, d0, d1);
    const int n = lane & 15;
    const int mb = (lane >> 4) << 3;
    if (n < NOUT) {
      const float bn = sB3[n];
#pragma unroll
      for (int half = 0; half < 2; ++half) {
        const v8f& d = half ? d1 : d0;
        const int rbase = half * 16 + mb;
#pragma unroll
        for (int i = 0; i < 8; ++i) {
          const int node = n0 + rbase + i;
          if (node < nNodes) xout[(size_t)node * NOUT + n] = d[i] + bn;
        }
      }
    }
  }
}

// ---------------------------------------------------------------------------
// R2 final edge kernel: [x[dst] | x[src] | E] (27) -> MLP -> sigmoid scalar
// ---------------------------------------------------------------------------
__global__ __launch_bounds__(128) void r2_kernel(
    const float* __restrict__ xin, const int* __restrict__ srcIdx,
    const int* __restrict__ dstIdx, const _Float16* __restrict__ eattr,
    const unsigned char* __restrict__ wp, float* __restrict__ out, int nEdges) {
  constexpr int Ff = 9, DIN = 27;
  DECLARE_MLP_LDS();
  __shared__ _Float16 sA[4][2][16 * 64];
  __shared__ int sDst[4][32];
  LOAD_MLP_LDS(wp);

  const int lane = threadIdx.x & 31;
  const int wave = threadIdx.x >> 5;
  _Float16* A0 = sA[wave][0];
  _Float16* A1 = sA[wave][1];
  int* myDst = sDst[wave];
  Frags F;
  BUILD_B_FRAGS(F);

  const int nPairs = (nEdges + 31) >> 5;
  for (int pair = blockIdx.x * 4 + wave; pair < nPairs; pair += gridDim.x * 4) {
    const int e0 = pair << 5;
#pragma unroll
    for (int i = 0; i < 4; ++i) {
      ((int4*)A0)[lane + 32 * i] = make_int4(0, 0, 0, 0);
      ((int4*)A1)[lane + 32 * i] = make_int4(0, 0, 0, 0);
    }
    {
      const int e = e0 + lane;
      myDst[lane] = (e < nEdges) ? dstIdx[e] : 0;
    }
    LDS_FENCE();
    for (int idx = lane; idx < 32 * DIN; idx += 32) {
      const int r = idx / DIN, c = idx - r * DIN;
      const int e = e0 + r;
      float v = 0.0f;
      if (e < nEdges) {
        if (c < Ff)           v = xin[(size_t)myDst[r] * Ff + c];
        else if (c < 2 * Ff)  v = xin[(size_t)srcIdx[e] * Ff + (c - Ff)];
        else                  v = (float)eattr[(size_t)e * Ff + (c - 2 * Ff)];
      }
      _Float16* A = (r < 16) ? A0 : A1;
      A[(r & 15) * 64 + c] = (_Float16)v;
    }
    LDS_FENCE();
    v8f d0, d1;
    mlp_tile2(A0, A1, F, sB1, sB2, d0, d1);
    const int n = lane & 15;
    const int mb = (lane >> 4) << 3;
    if (n == 0) {  // dout == 1: only column 0
      const float bn = sB3[0];
#pragma unroll
      for (int half = 0; half < 2; ++half) {
        const v8f& d = half ? d1 : d0;
        const int rbase = half * 16 + mb;
#pragma unroll
        for (int i = 0; i < 8; ++i) {
          const int e = e0 + rbase + i;
          if (e < nEdges) {
            const float v = d[i] + bn;
            out[e] = 1.0f / (1.0f + __expf(-v));
          }
        }
      }
    }
  }
}

// ---------------------------------------------------------------------------
// Host launch
// ---------------------------------------------------------------------------
extern "C" void kernel_launch(void* const* d_in, const int* in_sizes, int n_in,
                              void* d_out, int out_size, void* d_ws, size_t ws_size,
                              hipStream_t stream) {
  const float* x0  = (const float*)d_in[0];
  const int*   ei  = (const int*)d_in[1];
  const float* ea0 = (const float*)d_in[2];
  const int nNodes = in_sizes[0] / 3;
  const int nEdges = in_sizes[1] / 2;
  const int* src = ei;            // edge_index[0]
  const int* dst = ei + nEdges;   // edge_index[1]

  WPtrs wptrs;
  for (int i = 0; i < 42; ++i) wptrs.p[i] = (const float*)d_in[3 + i];

  unsigned char* ws = (unsigned char*)d_ws;
  size_t off = 0;
  auto take = [&](size_t bytes) -> unsigned char* {
    unsigned char* p = ws + off;
    off = (off + bytes + 255) & ~(size_t)255;
    return p;
  };
  unsigned char* wpack = take((size_t)7 * WPACK_STRIDE);
  float* xa   = (float*)take((size_t)nNodes * 9 * sizeof(float));
  float* xb   = (float*)take((size_t)nNodes * 9 * sizeof(float));
  float* aggr = (float*)take((size_t)nNodes * 9 * sizeof(float));
  _Float16* Ea = (_Float16*)take((size_t)nEdges * 9 * sizeof(_Float16));
  _Float16* Eb = (_Float16*)take((size_t)nEdges * 9 * sizeof(_Float16));
  (void)ws_size; (void)n_in; (void)out_size;

  pack_weights<<<dim3(1), dim3(256), 0, stream>>>(wptrs, wpack);

  const dim3 eGrid(2048), nGrid(512), blk(128);
  const size_t aggrBytes = (size_t)nNodes * 9 * sizeof(float);

  // Layer 1
  hipMemsetAsync(aggr, 0, aggrBytes, stream);
  r1_kernel<3, 3, 6, float><<<eGrid, blk, 0, stream>>>(
      x0, src, dst, ea0, wpack + 0 * WPACK_STRIDE, Ea, aggr, nEdges);
  o_kernel<3, 6, 6><<<nGrid, blk, 0, stream>>>(
      x0, aggr, wpack + 1 * WPACK_STRIDE, xa, nNodes);

  // Layer 2
  hipMemsetAsync(aggr, 0, aggrBytes, stream);
  r1_kernel<6, 6, 8, _Float16><<<eGrid, blk, 0, stream>>>(
      xa, src, dst, Ea, wpack + 2 * WPACK_STRIDE, Eb, aggr, nEdges);
  o_kernel<6, 8, 8><<<nGrid, blk, 0, stream>>>(
      xa, aggr, wpack + 3 * WPACK_STRIDE, xb, nNodes);

  // Layer 3
  hipMemsetAsync(aggr, 0, aggrBytes, stream);
  r1_kernel<8, 8, 9, _Float16><<<eGrid, blk, 0, stream>>>(
      xb, src, dst, Eb, wpack + 4 * WPACK_STRIDE, Ea, aggr, nEdges);
  o_kernel<8, 9, 9><<<nGrid, blk, 0, stream>>>(
      xb, aggr, wpack + 5 * WPACK_STRIDE, xa, nNodes);

  // Final R2 + sigmoid
  r2_kernel<<<eGrid, blk, 0, stream>>>(
      xa, src, dst, Ea, wpack + 6 * WPACK_STRIDE, (float*)d_out, nEdges);
}
